// StandardAttention_23167053594753
// MI455X (gfx1250) — matrix operand order repaired and checked by reading, not compiled, not run on hardware
//
#include <hip/hip_runtime.h>
#include <hip/hip_bf16.h>

// ---------------------------------------------------------------------------
// MI455X (gfx1250) multi-head attention forward, fp16 WMMA + fp32 accumulate.
// B=2, T=2048, D=1024, H=16, d=64.
// ---------------------------------------------------------------------------

typedef __attribute__((ext_vector_type(16))) _Float16 v16h;
typedef __attribute__((ext_vector_type(8)))  float    v8f;
typedef __attribute__((ext_vector_type(2)))  __fp16   v2fp16;

union FragH { v16h h; unsigned int u[8]; };

static __device__ __forceinline__ unsigned short f2h_bits(float x) {
    union { _Float16 h; unsigned short u; } cv;
    cv.h = (_Float16)x;
    return cv.u;
}

// Pack two fp32 -> two f16 in one v_cvt_pk_rtz_f16_f32
static __device__ __forceinline__ unsigned int pkh2(float a, float b) {
    union { v2fp16 h; unsigned int u; } cv;
    cv.h = __builtin_amdgcn_cvt_pkrtz(a, b);
    return cv.u;
}

#define WMMA_F16(A, B, C) \
    __builtin_amdgcn_wmma_f32_16x16x32_f16(false, (A), false, (B), (short)0, (C), false, false)

// Load an 8-uint fragment as two 16-byte chunks (global_load_b128 / ds_load_b128)
static __device__ __forceinline__ FragH load_frag2(const unsigned short* p0,
                                                   const unsigned short* p1) {
    FragH f;
    uint4 a = *(const uint4*)p0;
    uint4 b = *(const uint4*)p1;
    f.u[0] = a.x; f.u[1] = a.y; f.u[2] = a.z; f.u[3] = a.w;
    f.u[4] = b.x; f.u[5] = b.y; f.u[6] = b.z; f.u[7] = b.w;
    return f;
}

// A-fragment (16-bit A 16x32, ISA 7.12.2): per lane the 8 packed K-pairs live at
// ushort offsets {half*8 + 0..7} and {16 + half*8 + 0..7} within the 32-K row.
static __device__ __forceinline__ FragH load_a_frag(const unsigned short* row, int half) {
    return load_frag2(row + half * 8, row + 16 + half * 8);
}
// B-fragment (32x16): per lane (column) the K-pairs live at ushort offsets
// {half*8 + 0..7} and {16 + half*8 + 0..7}, matching the A-fragment.
static __device__ __forceinline__ FragH load_b_frag(const unsigned short* col, int half) {
    return load_frag2(col + half * 8, col + 16 + half * 8);
}

// ---------------------------------------------------------------------------
// Stage 1: Y = X @ W^T  (X fp32 [4096,1024], W fp32 [1024,1024]) -> f16.
//   z==0: Q (scaled by 1/8) -> [B,H,T,64]
//   z==1: K                 -> [B,H,T,64]
//   z==2: V (transposed)    -> [B,H,64,T]
// Block: 128 thr (4 waves). Block tile 64x64; each wave does 16 rows x 64 cols.
// Two-stage software pipeline: global loads for tile k+1 overlap WMMA on tile k.
// ---------------------------------------------------------------------------
__global__ __launch_bounds__(128) void qkv_gemm_kernel(
    const float* __restrict__ X,
    const float* __restrict__ Wq, const float* __restrict__ Wk,
    const float* __restrict__ Wv,
    unsigned short* __restrict__ qout,
    unsigned short* __restrict__ kout,
    unsigned short* __restrict__ vout)
{
    const int which = blockIdx.z;
    const float* __restrict__ W = (which == 0) ? Wq : ((which == 1) ? Wk : Wv);

    const int n0   = blockIdx.x * 64;
    const int m0   = blockIdx.y * 64;
    const int tid  = threadIdx.x;
    const int lane = tid & 31;
    const int wave = tid >> 5;
    const int half = lane >> 4;
    const int lr   = lane & 15;

    __shared__ unsigned short Xl[64 * 32];
    __shared__ unsigned short Wl[64 * 32];

    v8f acc[4] = {};

    // Tile = 64 rows x 32 K = 512 float4; 4 float4 per thread per matrix.
    float4 xr[4], wr[4];
    #pragma unroll
    for (int i = 0; i < 4; ++i) {                  // prologue loads (kk = 0)
        int e = tid + i * 128, r = e >> 3, c4 = e & 7;
        xr[i] = *(const float4*)&X[(size_t)(m0 + r) * 1024 + c4 * 4];
        wr[i] = *(const float4*)&W[(size_t)(n0 + r) * 1024 + c4 * 4];
    }

    for (int kk = 0; kk < 1024; kk += 32) {
        // Convert + store current tile registers into LDS (packed 8B stores)
        #pragma unroll
        for (int i = 0; i < 4; ++i) {
            int e = tid + i * 128;
            ((uint2*)Xl)[e] = make_uint2(pkh2(xr[i].x, xr[i].y), pkh2(xr[i].z, xr[i].w));
            ((uint2*)Wl)[e] = make_uint2(pkh2(wr[i].x, wr[i].y), pkh2(wr[i].z, wr[i].w));
        }
        __syncthreads();

        // Kick off next tile's global loads; they stay in flight across WMMAs
        if (kk + 32 < 1024) {
            #pragma unroll
            for (int i = 0; i < 4; ++i) {
                int e = tid + i * 128, r = e >> 3, c4 = e & 7;
                xr[i] = *(const float4*)&X[(size_t)(m0 + r) * 1024 + kk + 32 + c4 * 4];
                wr[i] = *(const float4*)&W[(size_t)(n0 + r) * 1024 + kk + 32 + c4 * 4];
            }
        }

        FragH a = load_a_frag(&Xl[((wave << 4) + lr) * 32], half);
        #pragma unroll
        for (int nt = 0; nt < 4; ++nt) {
            FragH b = load_b_frag(&Wl[((nt << 4) + lr) * 32], half);
            acc[nt] = WMMA_F16(a.h, b.h, acc[nt]);
        }
        __syncthreads();
    }

    // D layout: VGPR r -> row (r + 8*half), column = lr
    const int T = 2048;
    #pragma unroll
    for (int nt = 0; nt < 4; ++nt) {
        #pragma unroll
        for (int r = 0; r < 8; ++r) {
            int m = m0 + (wave << 4) + r + 8 * half;
            int n = n0 + (nt << 4) + lr;
            int b  = m >> 11, t  = m & 2047;
            int hh = n >> 6,  dd = n & 63;
            float val = acc[nt][r];
            if (which == 0)
                qout[(((b * 16 + hh) * T) + t) * 64 + dd] = f2h_bits(val * 0.125f);
            else if (which == 1)
                kout[(((b * 16 + hh) * T) + t) * 64 + dd] = f2h_bits(val);
            else
                vout[(((b * 16 + hh) * 64) + dd) * T + t] = f2h_bits(val);
        }
    }
}

// ---------------------------------------------------------------------------
// Stage 2: flash attention per (b,h). Block = 128 thr (4 waves), 64 queries.
// Each wave: 16 query rows, loops over 64-key tiles with online softmax.
// ---------------------------------------------------------------------------
__global__ __launch_bounds__(128) void flash_attn_kernel(
    const unsigned short* __restrict__ q,   // [B,H,T,64] f16, pre-scaled
    const unsigned short* __restrict__ k,   // [B,H,T,64] f16
    const unsigned short* __restrict__ vt,  // [B,H,64,T] f16 (transposed)
    unsigned short* __restrict__ y)         // [B,T,1024] f16 merged heads
{
    const int T = 2048;
    const int qt = blockIdx.x, h = blockIdx.y, b = blockIdx.z;
    const int tid = threadIdx.x, lane = tid & 31, wave = tid >> 5;
    const int half = lane >> 4, lr = lane & 15;

    const unsigned short* __restrict__ qh = q  + (size_t)((b * 16 + h) * T) * 64;
    const unsigned short* __restrict__ kh = k  + (size_t)((b * 16 + h) * T) * 64;
    const unsigned short* __restrict__ vh = vt + (size_t)((b * 16 + h) * 64) * T;

    __shared__ unsigned short Pl[4][16 * 64];   // per-wave P tile

    // Q fragments (16x64 = two K=32 chunks), resident for whole kernel
    FragH qa[2];
    const int qrow = qt * 64 + wave * 16 + lr;
    #pragma unroll
    for (int c = 0; c < 2; ++c)
        qa[c] = load_a_frag(&qh[qrow * 64 + c * 32], half);

    v8f accy[4] = {};
    float m_i[8], l_i[8];
    #pragma unroll
    for (int r = 0; r < 8; ++r) { m_i[r] = -1e30f; l_i[r] = 0.0f; }

    for (int kt = 0; kt < T; kt += 64) {
        // ---- S = Q @ K^T for this 16x64 tile (8 WMMAs) ----
        v8f s[4];
        #pragma unroll
        for (int ns = 0; ns < 4; ++ns) {
            v8f sacc = {};
            #pragma unroll
            for (int c = 0; c < 2; ++c) {
                // B[k][n] = K[key=n][dim=k]; a key's dims are contiguous
                FragH kb = load_b_frag(&kh[(kt + ns * 16 + lr) * 64 + c * 32], half);
                sacc = WMMA_F16(qa[c].h, kb.h, sacc);
            }
            s[ns] = sacc;
        }

        // ---- online softmax (row = VGPR index r + 8*half, 16 cols/lane-half)
        float mnew[8];
        #pragma unroll
        for (int r = 0; r < 8; ++r) {
            float t = fmaxf(fmaxf(s[0][r], s[1][r]), fmaxf(s[2][r], s[3][r]));
            #pragma unroll
            for (int off = 1; off < 16; off <<= 1)
                t = fmaxf(t, __shfl_xor(t, off, 32));
            mnew[r] = fmaxf(m_i[r], t);
        }
        float alpha[8];
        #pragma unroll
        for (int r = 0; r < 8; ++r) alpha[r] = __expf(m_i[r] - mnew[r]);
        #pragma unroll
        for (int ns = 0; ns < 4; ++ns)
            #pragma unroll
            for (int r = 0; r < 8; ++r) accy[ns][r] *= alpha[r];

        float rs[8];
        #pragma unroll
        for (int r = 0; r < 8; ++r) rs[r] = 0.0f;
        #pragma unroll
        for (int ns = 0; ns < 4; ++ns)
            #pragma unroll
            for (int r = 0; r < 8; ++r) {
                float p = __expf(s[ns][r] - mnew[r]);
                s[ns][r] = p;
                rs[r] += p;
            }
        #pragma unroll
        for (int r = 0; r < 8; ++r) {
            float t = rs[r];
            #pragma unroll
            for (int off = 1; off < 16; off <<= 1)
                t += __shfl_xor(t, off, 32);
            l_i[r] = l_i[r] * alpha[r] + t;
            m_i[r] = mnew[r];
        }

        // ---- D-layout -> A-layout for P via per-wave LDS tile ----
        __syncthreads();
        #pragma unroll
        for (int ns = 0; ns < 4; ++ns)
            #pragma unroll
            for (int r = 0; r < 8; ++r)
                Pl[wave][(r + 8 * half) * 64 + ns * 16 + lr] = f2h_bits(s[ns][r]);
        __syncthreads();

        FragH pa[2];
        #pragma unroll
        for (int c = 0; c < 2; ++c)
            pa[c] = load_a_frag(&Pl[wave][lr * 64 + c * 32], half);

        // ---- Y += P @ V (8 WMMAs); V^T layout gives contiguous key-pairs ----
        #pragma unroll
        for (int ns = 0; ns < 4; ++ns) {
            #pragma unroll
            for (int c = 0; c < 2; ++c) {
                FragH vb = load_b_frag(&vh[(ns * 16 + lr) * T + kt + c * 32], half);
                accy[ns] = WMMA_F16(pa[c].h, vb.h, accy[ns]);
            }
        }
    }

    // ---- normalize and store merged-head f16 y ----
    #pragma unroll
    for (int ns = 0; ns < 4; ++ns) {
        #pragma unroll
        for (int r = 0; r < 8; ++r) {
            int t = qt * 64 + wave * 16 + r + 8 * half;
            int n = h * 64 + ns * 16 + lr;
            y[(size_t)(b * T + t) * 1024 + n] = f2h_bits(accy[ns][r] / l_i[r]);
        }
    }
}

// ---------------------------------------------------------------------------
// Stage 3: out = Y(f16) @ Wo^T -> fp32 [4096,1024], same pipelined skeleton.
// ---------------------------------------------------------------------------
__global__ __launch_bounds__(128) void out_gemm_kernel(
    const unsigned short* __restrict__ Y,
    const float* __restrict__ Wo,
    float* __restrict__ out)
{
    const int n0   = blockIdx.x * 64;
    const int m0   = blockIdx.y * 64;
    const int tid  = threadIdx.x;
    const int lane = tid & 31;
    const int wave = tid >> 5;
    const int half = lane >> 4;
    const int lr   = lane & 15;

    __shared__ unsigned short Xl[64 * 32];
    __shared__ unsigned short Wl[64 * 32];

    v8f acc[4] = {};

    // Y tile: 2048 ushorts = 256 uint4 -> 2 per thread. Wo tile: 4 float4/thread.
    uint4  yr[2];
    float4 wr[4];
    #pragma unroll
    for (int i = 0; i < 2; ++i) {
        int e = tid + i * 128, r = e >> 2, c = (e & 3) * 8;
        yr[i] = *(const uint4*)&Y[(size_t)(m0 + r) * 1024 + c];
    }
    #pragma unroll
    for (int i = 0; i < 4; ++i) {
        int e = tid + i * 128, r = e >> 3, c4 = e & 7;
        wr[i] = *(const float4*)&Wo[(size_t)(n0 + r) * 1024 + c4 * 4];
    }

    for (int kk = 0; kk < 1024; kk += 32) {
        #pragma unroll
        for (int i = 0; i < 2; ++i)
            ((uint4*)Xl)[tid + i * 128] = yr[i];
        #pragma unroll
        for (int i = 0; i < 4; ++i)
            ((uint2*)Wl)[tid + i * 128] =
                make_uint2(pkh2(wr[i].x, wr[i].y), pkh2(wr[i].z, wr[i].w));
        __syncthreads();

        if (kk + 32 < 1024) {
            #pragma unroll
            for (int i = 0; i < 2; ++i) {
                int e = tid + i * 128, r = e >> 2, c = (e & 3) * 8;
                yr[i] = *(const uint4*)&Y[(size_t)(m0 + r) * 1024 + kk + 32 + c];
            }
            #pragma unroll
            for (int i = 0; i < 4; ++i) {
                int e = tid + i * 128, r = e >> 3, c4 = e & 7;
                wr[i] = *(const float4*)&Wo[(size_t)(n0 + r) * 1024 + kk + 32 + c4 * 4];
            }
        }

        FragH a = load_a_frag(&Xl[((wave << 4) + lr) * 32], half);
        #pragma unroll
        for (int nt = 0; nt < 4; ++nt) {
            FragH b = load_b_frag(&Wl[((nt << 4) + lr) * 32], half);
            acc[nt] = WMMA_F16(a.h, b.h, acc[nt]);
        }
        __syncthreads();
    }

    #pragma unroll
    for (int nt = 0; nt < 4; ++nt) {
        #pragma unroll
        for (int r = 0; r < 8; ++r) {
            int m = m0 + (wave << 4) + r + 8 * half;
            int n = n0 + (nt << 4) + lr;
            out[(size_t)m * 1024 + n] = acc[nt][r];
        }
    }
}

// ---------------------------------------------------------------------------
extern "C" void kernel_launch(void* const* d_in, const int* in_sizes, int n_in,
                              void* d_out, int out_size, void* d_ws, size_t ws_size,
                              hipStream_t stream) {
    (void)in_sizes; (void)n_in; (void)out_size; (void)ws_size;

    const float* x  = (const float*)d_in[0];
    const float* Wq = (const float*)d_in[1];
    const float* Wk = (const float*)d_in[2];
    const float* Wv = (const float*)d_in[3];
    const float* Wo = (const float*)d_in[4];
    float* out = (float*)d_out;

    const size_t MN = (size_t)4096 * 1024;
    unsigned short* qf = (unsigned short*)d_ws;   // f16 [B,H,T,64] (pre-scaled)
    unsigned short* kf = qf + MN;                 // f16 [B,H,T,64]
    unsigned short* vf = kf + MN;                 // f16 [B,H,64,T] (transposed)
    unsigned short* yf = vf + MN;                 // f16 [B,T,1024]

    dim3 g1(1024 / 64, 4096 / 64, 3);             // 16 x 64 x {q,k,v}
    qkv_gemm_kernel<<<g1, 128, 0, stream>>>(x, Wq, Wk, Wv, qf, kf, vf);

    dim3 g2(2048 / 64, 16, 2);                    // q-tiles x H x B
    flash_attn_kernel<<<g2, 128, 0, stream>>>(qf, kf, vf, yf);

    dim3 g3(1024 / 64, 4096 / 64);
    out_gemm_kernel<<<g3, 128, 0, stream>>>(yf, Wo, out);
}
